// ApiAdapter_5351529251293
// MI455X (gfx1250) — matrix operand order repaired and checked by reading, not compiled, not run on hardware
//
#include <hip/hip_runtime.h>
#include <hip/hip_bf16.h>
#include <math.h>

// ---------------------------------------------------------------------------
// MI455X (gfx1250) implementation of the Mamba-2 memory-compression reference.
// GEMMs (~240 GFLOP hot spot) run on v_wmma_f32_16x16x32_f16 with f32
// accumulate.  LDS tiles are stored in WMMA *fragment order* so each lane's
// 16-f16 operand is two ds_load_b128.  Each wave owns a 32x64 C tile
// (2 A fragments x 4 B fragments = 8 WMMA per K-step, 12 ds_load_b128).
// ---------------------------------------------------------------------------

typedef __attribute__((ext_vector_type(16))) _Float16 v16h;
typedef __attribute__((ext_vector_type(8)))  _Float16 v8h;
typedef __attribute__((ext_vector_type(8)))  float    v8f;

#define DM     1024   // D_MODEL
#define NL     2      // N_LAYERS
#define DSTATE 64
#define DHEAD  64
#define DI     2048   // D_INNER
#define NH     32     // NHEADS
#define CONVD  2176   // D_INNER + 2*D_STATE
#define DIN    4256   // 2*D_INNER + 2*D_STATE + NHEADS
#define KC     4      // KCONV
#define PATCH  32
#define LSEQ   1024
#define MMEM   8
#define TTOT   9216   // (1+M)*L
#define EPSF   1e-5f

__device__ __forceinline__ float siluf(float x) { return x / (1.f + __expf(-x)); }
__device__ __forceinline__ float softplusf(float x) {
  return (x > 20.f) ? x : log1pf(__expf(x));
}

// --------------------------- sequence build --------------------------------

__global__ __launch_bounds__(256) void k_fill_seq(const float* __restrict__ q,
                                                  float* __restrict__ seq) {
  size_t i = (size_t)blockIdx.x * 256 + threadIdx.x;
  size_t t = i / DM;
  seq[i] = (t < LSEQ) ? q[i] : 0.f;
}

// per-row std (ddof=1) over D columns
__global__ __launch_bounds__(256) void k_row_std(const float* __restrict__ x,
                                                 int D, float* __restrict__ ent) {
  int row = blockIdx.x;
  const float* p = x + (size_t)row * D;
  float s = 0.f, s2 = 0.f;
  for (int i = threadIdx.x; i < D; i += 256) { float v = p[i]; s += v; s2 += v * v; }
  __shared__ float sh1[256], sh2[256];
  sh1[threadIdx.x] = s; sh2[threadIdx.x] = s2; __syncthreads();
  for (int st = 128; st > 0; st >>= 1) {
    if (threadIdx.x < st) { sh1[threadIdx.x] += sh1[threadIdx.x + st];
                            sh2[threadIdx.x] += sh2[threadIdx.x + st]; }
    __syncthreads();
  }
  if (threadIdx.x == 0) {
    float su = sh1[0], sq = sh2[0];
    float var = (sq - su * su / (float)D) / (float)(D - 1);
    ent[row] = sqrtf(fmaxf(var, 0.f));
  }
}

// top-PATCH indices of ent[blockIdx.x*LSEQ .. +LSEQ), sorted ascending
__global__ __launch_bounds__(256) void k_topk(const float* __restrict__ ent,
                                              int* __restrict__ idx_out) {
  const float* e = ent + (size_t)blockIdx.x * LSEQ;
  int* out = idx_out + blockIdx.x * PATCH;
  __shared__ float v[LSEQ];
  __shared__ unsigned char tk[LSEQ];
  __shared__ float rv[256];
  __shared__ int   ri[256];
  __shared__ int   sel[PATCH];
  for (int i = threadIdx.x; i < LSEQ; i += 256) { v[i] = e[i]; tk[i] = 0; }
  __syncthreads();
  for (int k = 0; k < PATCH; ++k) {
    float bv = -1e30f; int bi = LSEQ;
    for (int i = threadIdx.x; i < LSEQ; i += 256)
      if (!tk[i] && (v[i] > bv || (v[i] == bv && i < bi))) { bv = v[i]; bi = i; }
    rv[threadIdx.x] = bv; ri[threadIdx.x] = bi;
    __syncthreads();
    for (int s = 128; s > 0; s >>= 1) {
      if (threadIdx.x < s) {
        float ov = rv[threadIdx.x + s]; int oi = ri[threadIdx.x + s];
        if (ov > rv[threadIdx.x] || (ov == rv[threadIdx.x] && oi < ri[threadIdx.x])) {
          rv[threadIdx.x] = ov; ri[threadIdx.x] = oi;
        }
      }
      __syncthreads();
    }
    if (threadIdx.x == 0) { sel[k] = ri[0]; tk[ri[0]] = 1; }
    __syncthreads();
  }
  if (threadIdx.x == 0) {
    for (int a = 1; a < PATCH; ++a) {
      int key = sel[a], b = a - 1;
      while (b >= 0 && sel[b] > key) { sel[b + 1] = sel[b]; --b; }
      sel[b + 1] = key;
    }
    for (int a = 0; a < PATCH; ++a) out[a] = sel[a];
  }
}

// scatter int8-quantized memory rows into seq (base==0 so diff==mem)
__global__ __launch_bounds__(256) void k_scatter(const float* __restrict__ mems,
                                                 const int* __restrict__ idxb,
                                                 float* __restrict__ seq) {
  int m = blockIdx.x, k = blockIdx.y;
  int row = idxb[m * PATCH + k];
  const float* src = mems + ((size_t)m * LSEQ + row) * DM;
  float* dst = seq + ((size_t)(1 + m) * LSEQ + row) * DM;
  #pragma unroll
  for (int i = 0; i < 4; ++i) {
    int d = threadIdx.x + i * 256;
    dst[d] = (float)(signed char)(int)src[d];   // float -> int8 -> float
  }
}

// ------------------------------- norms -------------------------------------

__global__ __launch_bounds__(256) void k_rmsnorm(const float* __restrict__ x,
                                                 const float* __restrict__ w,
                                                 float* __restrict__ y) {
  int t = blockIdx.x;
  const float* p = x + (size_t)t * DM;
  float v[4]; float ss = 0.f;
  #pragma unroll
  for (int i = 0; i < 4; ++i) { v[i] = p[threadIdx.x + i * 256]; ss += v[i] * v[i]; }
  __shared__ float sh[256];
  sh[threadIdx.x] = ss; __syncthreads();
  for (int s = 128; s > 0; s >>= 1) {
    if (threadIdx.x < s) sh[threadIdx.x] += sh[threadIdx.x + s];
    __syncthreads();
  }
  float scale = rsqrtf(sh[0] / (float)DM + EPSF);
  float* q = y + (size_t)t * DM;
  #pragma unroll
  for (int i = 0; i < 4; ++i) { int d = threadIdx.x + i * 256; q[d] = v[i] * scale * w[d]; }
}

// y <- rmsnorm(y * silu(z)) * nw   (in place over ybuf)
__global__ __launch_bounds__(256) void k_gated_rmsnorm(float* __restrict__ y,
                                                       const float* __restrict__ zx,
                                                       const float* __restrict__ nw) {
  int t = blockIdx.x;
  float* yp = y + (size_t)t * DI;
  const float* zp = zx + (size_t)t * DIN;   // z = first D_INNER cols of zxbcdt
  float g[8]; float ss = 0.f;
  #pragma unroll
  for (int i = 0; i < 8; ++i) {
    int d = threadIdx.x + i * 256;
    float gv = yp[d] * siluf(zp[d]);
    g[i] = gv; ss += gv * gv;
  }
  __shared__ float sh[256];
  sh[threadIdx.x] = ss; __syncthreads();
  for (int s = 128; s > 0; s >>= 1) {
    if (threadIdx.x < s) sh[threadIdx.x] += sh[threadIdx.x + s];
    __syncthreads();
  }
  float scale = rsqrtf(sh[0] / (float)DI + EPSF);
  #pragma unroll
  for (int i = 0; i < 8; ++i) { int d = threadIdx.x + i * 256; yp[d] = g[i] * scale * nw[d]; }
}

// ------------------------------ WMMA GEMM ----------------------------------
// C[T x N] = A[T x K] @ B[K x N] (+bias) (+resid).  f32 in memory, f16 in LDS.
// Block tile 256x64, K-step 32, 8 waves; each wave owns a 32x64 C tile:
// 2 A fragments x 4 B fragments -> 8 v_wmma per K-step, 12 ds_load_b128.
// LDS tiles are in WMMA *fragment order*:
//   Afrag[(rowblk*32 + lane)*2 + half] : v8h (16 KB), rowblk = 0..15
//   Bfrag[(ns*32    + lane)*2 + half]  : v8h ( 4 KB), ns     = 0..3
// Fragment <-> global mapping (ISA 7.12.2):
//   A: lane = mrow + 16*hi; half h covers K = 8*(hi + 2*h) .. +7 (contiguous)
//   B: lane = ncol + 16*hi; half h covers K = 8*(hi + 2*h) .. +7 (stride N)
#define BM 256
#define BN 64
#define BK 32

__global__ __launch_bounds__(256) void k_gemm_wmma(const float* __restrict__ A,
                                                   const float* __restrict__ B,
                                                   const float* __restrict__ bias,
                                                   const float* __restrict__ resid,
                                                   float* __restrict__ C,
                                                   int K, int N) {
  __shared__ v8h Afrag[16 * 32 * 2];  // 16 row-blocks x 32 lanes x 2 halves
  __shared__ v8h Bfrag[4 * 32 * 2];   //  4 n-subtiles x 32 lanes x 2 halves
  const int tid  = threadIdx.x;
  const int lane = tid & 31;
  const int w    = tid >> 5;
  const int m0   = blockIdx.y * BM;
  const int n0   = blockIdx.x * BN;

  const v8f vzero = {0.f, 0.f, 0.f, 0.f, 0.f, 0.f, 0.f, 0.f};
  v8f acc[2][4] = {{vzero, vzero, vzero, vzero}, {vzero, vzero, vzero, vzero}};

  const int ksteps = K / BK;
  for (int kt = 0; kt < ksteps; ++kt) {
    const int kb = kt * BK;

    // ---- stage A tile (256x32) directly into fragment order ----
    #pragma unroll
    for (int gi = 0; gi < 4; ++gi) {
      int g     = tid + gi * 256;          // 0..1023 groups of 8
      int rb    = g >> 6;                  // row-block 0..15
      int rem   = g & 63;
      int slane = rem & 31;
      int h     = rem >> 5;
      int mrow  = slane & 15;
      int shi   = slane >> 4;
      int K0    = 8 * (shi + 2 * h);       // contiguous 8 K values
      const float4* src = (const float4*)(A + (size_t)(m0 + rb * 16 + mrow) * K + kb + K0);
      float4 u0 = src[0], u1 = src[1];
      v8h a;
      a[0] = (_Float16)u0.x; a[1] = (_Float16)u0.y;
      a[2] = (_Float16)u0.z; a[3] = (_Float16)u0.w;
      a[4] = (_Float16)u1.x; a[5] = (_Float16)u1.y;
      a[6] = (_Float16)u1.z; a[7] = (_Float16)u1.w;
      Afrag[(rb * 32 + slane) * 2 + h] = a;  // one ds_store_b128
    }
    // ---- stage B tile (32x64) directly into fragment order ----
    {
      int ns    = tid >> 6;                // 0..3
      int rem   = tid & 63;
      int blane = rem & 31;
      int bh    = rem >> 5;
      int bhi   = blane >> 4;
      int ncol  = n0 + ns * 16 + (blane & 15);
      int K0    = 8 * (bhi + 2 * bh);      // 8 K values, stride N in global
      v8h b;
      if (ncol < N) {
        const float* src = B + (size_t)(kb + K0) * N + ncol;
        #pragma unroll
        for (int j = 0; j < 8; ++j) b[j] = (_Float16)src[(size_t)j * N];
      } else {
        #pragma unroll
        for (int j = 0; j < 8; ++j) b[j] = (_Float16)0.f;
      }
      Bfrag[(ns * 32 + blane) * 2 + bh] = b;  // one ds_store_b128
    }
    if (kt + 1 < ksteps) {   // global_prefetch_b8 the next K tiles
      __builtin_prefetch(A + (size_t)(m0 + tid) * K + kb + BK, 0, 1);
      __builtin_prefetch(B + (size_t)(kb + BK + (tid >> 3)) * N + n0, 0, 1);
    }
    __syncthreads();

    // ---- fragments: 2 ds_load_b128 each, shuffle into v16h ----
    v8h a00 = Afrag[((w * 2 + 0) * 32 + lane) * 2 + 0];
    v8h a01 = Afrag[((w * 2 + 0) * 32 + lane) * 2 + 1];
    v8h a10 = Afrag[((w * 2 + 1) * 32 + lane) * 2 + 0];
    v8h a11 = Afrag[((w * 2 + 1) * 32 + lane) * 2 + 1];
    v16h af0 = __builtin_shufflevector(a00, a01, 0, 1, 2, 3, 4, 5, 6, 7,
                                                 8, 9, 10, 11, 12, 13, 14, 15);
    v16h af1 = __builtin_shufflevector(a10, a11, 0, 1, 2, 3, 4, 5, 6, 7,
                                                 8, 9, 10, 11, 12, 13, 14, 15);
    #pragma unroll
    for (int ns = 0; ns < 4; ++ns) {
      v8h b0 = Bfrag[(ns * 32 + lane) * 2 + 0];
      v8h b1 = Bfrag[(ns * 32 + lane) * 2 + 1];
      v16h bf = __builtin_shufflevector(b0, b1, 0, 1, 2, 3, 4, 5, 6, 7,
                                                8, 9, 10, 11, 12, 13, 14, 15);
      acc[0][ns] = __builtin_amdgcn_wmma_f32_16x16x32_f16(
          false, af0, false, bf, (short)0, acc[0][ns], false, false);
      acc[1][ns] = __builtin_amdgcn_wmma_f32_16x16x32_f16(
          false, af1, false, bf, (short)0, acc[1][ns], false, false);
    }
    __syncthreads();
  }

  // store: C/D layout — VGPR r holds M=r (lanes 0-15) / M=8+r (lanes 16-31)
  const int cl = lane & 15;
  #pragma unroll
  for (int rbi = 0; rbi < 2; ++rbi) {
    const int rbase = m0 + (w * 2 + rbi) * 16 + ((lane >> 4) << 3);
    #pragma unroll
    for (int ns = 0; ns < 4; ++ns) {
      int col = n0 + (ns << 4) + cl;
      if (col < N) {
        float bb = bias ? bias[col] : 0.f;
        #pragma unroll
        for (int r = 0; r < 8; ++r) {
          size_t off = (size_t)(rbase + r) * N + col;
          float v = acc[rbi][ns][r] + bb;
          if (resid) v += resid[off];
          C[off] = v;
        }
      }
    }
  }
}

// --------------------------- conv / dt / SSD -------------------------------

__global__ __launch_bounds__(256) void k_conv_silu(const float* __restrict__ zx,
                                                   const float* __restrict__ cw,
                                                   const float* __restrict__ cb,
                                                   float* __restrict__ xc) {
  int c = blockIdx.y * 256 + threadIdx.x;
  int t = blockIdx.x;
  if (c >= CONVD) return;
  float acc = cb[c];
  #pragma unroll
  for (int j = 0; j < KC; ++j) {
    int ts = t - (KC - 1) + j;
    if (ts >= 0) acc += cw[c * KC + j] * zx[(size_t)ts * DIN + DI + c];
  }
  xc[(size_t)t * CONVD + c] = siluf(acc);
}

__global__ __launch_bounds__(256) void k_dt(const float* __restrict__ zx,
                                            const float* __restrict__ dtb,
                                            float* __restrict__ dts) {
  int i = blockIdx.x * 256 + threadIdx.x;
  int t = i / NH, h = i % NH;
  float v = zx[(size_t)t * DIN + DI + CONVD + h] + dtb[h];
  dts[i] = softplusf(v);
}

// one workgroup per head; 64x64 state in registers (16 f32/thread)
__global__ __launch_bounds__(256) void k_ssd(const float* __restrict__ xc,
                                             const float* __restrict__ dts,
                                             const float* __restrict__ A_log,
                                             const float* __restrict__ Dsk,
                                             float* __restrict__ y) {
  const int h   = blockIdx.x;
  const int tid = threadIdx.x;
  const int p   = tid >> 2;   // 0..63 (head dim)
  const int nq  = tid & 3;    // 16-state chunk
  const float Av = -__expf(A_log[h]);
  const float Dv = Dsk[h];
  float hreg[16];
  #pragma unroll
  for (int i = 0; i < 16; ++i) hreg[i] = 0.f;
  __shared__ float xs[DHEAD], bs[DSTATE], cs[DSTATE];
  __shared__ float red[DHEAD][4];
  for (int t = 0; t < TTOT; ++t) {
    const float* row = xc + (size_t)t * CONVD;
    if (tid < 64)        xs[tid]       = row[h * DHEAD + tid];
    else if (tid < 128)  bs[tid - 64]  = row[DI + (tid - 64)];
    else if (tid < 192)  cs[tid - 128] = row[DI + DSTATE + (tid - 128)];
    __syncthreads();
    const float dt = dts[(size_t)t * NH + h];
    const float dA = __expf(dt * Av);
    const float dx = dt * xs[p];
    float part = 0.f;
    #pragma unroll
    for (int i = 0; i < 16; ++i) {
      int n = (nq << 4) + i;
      float hv = hreg[i] * dA + dx * bs[n];
      hreg[i] = hv;
      part += hv * cs[n];
    }
    red[p][nq] = part;
    __syncthreads();
    if (tid < 64) {
      float yv = red[tid][0] + red[tid][1] + red[tid][2] + red[tid][3] + Dv * xs[tid];
      y[(size_t)t * DI + h * DHEAD + tid] = yv;
    }
    __syncthreads();
  }
}

// ------------------------- aggregate / output ------------------------------

__global__ __launch_bounds__(256) void k_aggregate(const float* __restrict__ seq,
                                                   float* __restrict__ agg) {
  size_t i = (size_t)blockIdx.x * 256 + threadIdx.x;   // i < LSEQ*DM
  size_t t = i / DM, d = i % DM;
  float s = 0.f;
  #pragma unroll
  for (int r = 0; r < 1 + MMEM; ++r)
    s += seq[((size_t)r * LSEQ + t) * DM + d];
  agg[i] = s / (float)(1 + MMEM);
}

// d_out (float): [PATCH*DM patch-int8-values][PATCH indices][L*DM zeros (base)]
__global__ __launch_bounds__(256) void k_writeout(const float* __restrict__ agg,
                                                  const int* __restrict__ idx,
                                                  float* __restrict__ out) {
  size_t i = (size_t)blockIdx.x * 256 + threadIdx.x;
  const size_t np = (size_t)PATCH * DM;
  const size_t ni = np + PATCH;
  const size_t nt = ni + (size_t)LSEQ * DM;
  if (i < np) {
    int k = (int)(i / DM), d = (int)(i % DM);
    out[i] = (float)(signed char)(int)agg[(size_t)idx[k] * DM + d];
  } else if (i < ni) {
    out[i] = (float)idx[i - np];
  } else if (i < nt) {
    out[i] = 0.f;     // base is all-zeros
  }
}

// ------------------------------ launcher -----------------------------------

extern "C" void kernel_launch(void* const* d_in, const int* in_sizes, int n_in,
                              void* d_out, int out_size, void* d_ws, size_t ws_size,
                              hipStream_t stream) {
  const float* query   = (const float*)d_in[0];
  const float* mems    = (const float*)d_in[1];
  const float* pre_w   = (const float*)d_in[2];
  const float* in_w    = (const float*)d_in[3];
  const float* in_b    = (const float*)d_in[4];
  const float* conv_w  = (const float*)d_in[5];
  const float* conv_b  = (const float*)d_in[6];
  const float* dt_bias = (const float*)d_in[7];
  const float* A_log   = (const float*)d_in[8];
  const float* D_skip  = (const float*)d_in[9];
  const float* norm_w  = (const float*)d_in[10];
  const float* out_w   = (const float*)d_in[11];
  (void)in_sizes; (void)n_in; (void)out_size; (void)ws_size;

  float* ws = (float*)d_ws;
  size_t o = 0;
  float* seq  = ws + o; o += (size_t)TTOT * DM;     // 37.7 MB
  float* xn   = ws + o; o += (size_t)TTOT * DM;     // 37.7 MB
  float* zx   = ws + o; o += (size_t)TTOT * DIN;    // 156.9 MB
  float* xc   = ws + o; o += (size_t)TTOT * CONVD;  // 80.2 MB
  float* dts  = ws + o; o += (size_t)TTOT * NH;     // 1.2 MB
  float* ybuf = ws + o; o += (size_t)TTOT * DI;     // 75.5 MB
  float* agg  = ws + o; o += (size_t)LSEQ * DM;     // 4.2 MB
  float* ent  = ws + o; o += (size_t)MMEM * LSEQ;
  int*   idxb = (int*)(ws + o);                     // (M+1)*PATCH ints

  // 1) compression + sequence build (base == 0 so diff == memory)
  k_fill_seq<<<(TTOT * DM) / 256, 256, 0, stream>>>(query, seq);
  k_row_std<<<MMEM * LSEQ, 256, 0, stream>>>(mems, DM, ent);
  k_topk<<<MMEM, 256, 0, stream>>>(ent, idxb);
  k_scatter<<<dim3(MMEM, PATCH), 256, 0, stream>>>(mems, idxb, seq);

  // 2) two mixer layers
  for (int l = 0; l < NL; ++l) {
    k_rmsnorm<<<TTOT, 256, 0, stream>>>(seq, pre_w + l * DM, xn);
    dim3 g_in((DIN + BN - 1) / BN, TTOT / BM);      // 67 x 36
    k_gemm_wmma<<<g_in, 256, 0, stream>>>(xn, in_w + (size_t)l * DM * DIN,
                                          in_b + l * DIN, nullptr, zx, DM, DIN);
    dim3 g_cv(TTOT, (CONVD + 255) / 256);
    k_conv_silu<<<g_cv, 256, 0, stream>>>(zx, conv_w + (size_t)l * CONVD * KC,
                                          conv_b + l * CONVD, xc);
    k_dt<<<(TTOT * NH) / 256, 256, 0, stream>>>(zx, dt_bias + l * NH, dts);
    k_ssd<<<NH, 256, 0, stream>>>(xc, dts, A_log + l * NH, D_skip + l * NH, ybuf);
    k_gated_rmsnorm<<<TTOT, 256, 0, stream>>>(ybuf, zx, norm_w + l * DI);
    dim3 g_out(DM / BN, TTOT / BM);                 // 16 x 36
    k_gemm_wmma<<<g_out, 256, 0, stream>>>(ybuf, out_w + (size_t)l * DI * DM,
                                           nullptr, seq, seq, DI, DM);
  }

  // 3) aggregate + final compression + output
  k_aggregate<<<(LSEQ * DM) / 256, 256, 0, stream>>>(seq, agg);
  k_row_std<<<LSEQ, 256, 0, stream>>>(agg, DM, ent);
  k_topk<<<1, 256, 0, stream>>>(ent, idxb + MMEM * PATCH);
  size_t tot = (size_t)PATCH * DM + PATCH + (size_t)LSEQ * DM;
  k_writeout<<<(unsigned)((tot + 255) / 256), 256, 0, stream>>>(
      agg, idxb + MMEM * PATCH, (float*)d_out);
}
